// CausalCrossConditionalSelfAttention_57062935494869
// MI455X (gfx1250) — matrix operand order repaired and checked by reading, not compiled, not run on hardware
//
#include <hip/hip_runtime.h>

// ---------------- types ----------------
typedef __bf16 v16bf __attribute__((ext_vector_type(16)));
typedef __bf16 v8bf  __attribute__((ext_vector_type(8)));
typedef __bf16 v4bf  __attribute__((ext_vector_type(4)));
typedef float  v8f   __attribute__((ext_vector_type(8)));
typedef float  v4f   __attribute__((ext_vector_type(4)));

#define B_  4
#define T_  2048
#define C_  1024
#define H_  16
#define D_  64
#define M_  (B_ * T_)   // 8192 rows

__device__ __forceinline__ __bf16 f2bf(float f) {
    union { float f; unsigned u; } v; v.f = f;
    unsigned r = v.u + 0x7FFFu + ((v.u >> 16) & 1u);   // round-to-nearest-even
    union { unsigned short s; __bf16 b; } o; o.s = (unsigned short)(r >> 16);
    return o.b;
}

__device__ __forceinline__ v8f wmma_bf16(v16bf a, v16bf b, v8f c) {
    // D = A(16x32 bf16) * B(32x16 bf16) + C(16x16 f32)
    return __builtin_amdgcn_wmma_f32_16x16x32_bf16(false, a, false, b, (short)0, c, false, false);
}

// Async global->LDS 16B copy (ASYNCcnt path, bypasses VGPRs).
// lds_off: byte offset inside workgroup LDS allocation; base: uniform 64-bit
// global base (SGPR pair); goff: per-lane signed 32-bit byte offset.
__device__ __forceinline__ void async_b128(unsigned lds_off, unsigned goff,
                                           unsigned long long base) {
    asm volatile("global_load_async_to_lds_b128 %0, %1, %2"
                 :: "v"(lds_off), "v"(goff), "s"(base) : "memory");
}
__device__ __forceinline__ void wait_async0() {
    asm volatile("s_wait_asynccnt 0x0" ::: "memory");
}

// A-fragment (16x32, bf16): lane holds row (lane&15); K=0..7/16..23 for lanes
// 0-15, K=8..15/24..31 for lanes 16-31 (ISA 16-bit A layout).
__device__ __forceinline__ v16bf load_A_frag(const __bf16* base, int stride, int lane) {
    int m  = lane & 15;
    int kh = (lane >> 4) * 8;
    const __bf16* p = base + m * stride;
    v16bf a;
    *(v8bf*)&a         = *(const v8bf*)(p + kh);
    *(((v8bf*)&a) + 1) = *(const v8bf*)(p + 16 + kh);
    return a;
}

// B-fragment (32x16, bf16) from a tile stored as rows of the N index:
// lane holds column (lane&15); K=0..7/16..23 for lanes 0-15, K=8..15/24..31 for lanes 16-31.
__device__ __forceinline__ v16bf load_B_frag(const __bf16* base, int stride, int lane) {
    int n  = lane & 15;
    int kh = (lane >> 4) * 8;
    const __bf16* p = base + n * stride;
    v16bf b;
    *(v8bf*)&b         = *(const v8bf*)(p + kh);
    *(((v8bf*)&b) + 1) = *(const v8bf*)(p + 16 + kh);
    return b;
}

// ---------------- f32 -> bf16 convert (vectorized) ----------------
__global__ void cvt_f32_bf16_v4(const float* __restrict__ src,
                                __bf16* __restrict__ dst, int n4) {
    int i = blockIdx.x * blockDim.x + threadIdx.x;
    int stride = gridDim.x * blockDim.x;
    for (; i < n4; i += stride) {
        v4f x = ((const v4f*)src)[i];
        v4bf o;
        #pragma unroll
        for (int j = 0; j < 4; ++j) o[j] = f2bf(x[j]);
        ((v4bf*)dst)[i] = o;
    }
}

// ---------------- WMMA GEMM: Y = X(bf16)[M x 1024] * W^T + bias ----------------
// W is [out=1024][in=1024] row-major, so B[k][n] = W[n][k] -> stage W rows.
// MODE 0: f32 row-major [M x 1024] (projection -> d_out)
// MODE 1: bf16 scattered to [B,H,T,d] (QKV staging)
template <int MODE>
__global__ __launch_bounds__(256)
void gemm_bf16(const __bf16* __restrict__ A, const __bf16* __restrict__ W,
               const float* __restrict__ bias,
               float* __restrict__ outF, __bf16* __restrict__ outQ) {
    __shared__ __align__(64) __bf16 As[2][128][32];   // 2 x 8 KB (double buffer)
    __shared__ __align__(64) __bf16 Ws[2][128][32];   // 2 x 8 KB
    const int K = C_;
    const int m0 = blockIdx.y * 128;
    const int n0 = blockIdx.x * 128;
    const int tid  = threadIdx.x;
    const int w    = tid >> 5;
    const int lane = tid & 31;
    const int wm = (w & 3) * 32;   // wave row offset in block tile
    const int wn = (w >> 2) * 64;  // wave col offset in block tile

    const unsigned long long abase = (unsigned long long)A;
    const unsigned long long wbase = (unsigned long long)W;

    // per-thread staging chunks: 512 x 16B per array, 2 chunks each
    const int c0   = tid;
    const int c1   = tid + 256;
    const int row0 = c0 >> 2, col0 = (c0 & 3) * 8;
    const int row1 = c1 >> 2, col1 = (c1 & 3) * 8;

    auto issue = [&](int buf, int kk) {
        async_b128((unsigned)(size_t)&As[buf][row0][col0],
                   (unsigned)(((m0 + row0) * K + kk + col0) * 2), abase);
        async_b128((unsigned)(size_t)&As[buf][row1][col1],
                   (unsigned)(((m0 + row1) * K + kk + col1) * 2), abase);
        async_b128((unsigned)(size_t)&Ws[buf][row0][col0],
                   (unsigned)(((n0 + row0) * K + kk + col0) * 2), wbase);
        async_b128((unsigned)(size_t)&Ws[buf][row1][col1],
                   (unsigned)(((n0 + row1) * K + kk + col1) * 2), wbase);
    };

    const v8f vz = {0.f,0.f,0.f,0.f,0.f,0.f,0.f,0.f};
    v8f acc[2][4];
    #pragma unroll
    for (int mt = 0; mt < 2; ++mt)
        #pragma unroll
        for (int nt = 0; nt < 4; ++nt) acc[mt][nt] = vz;

    issue(0, 0);
    wait_async0();
    __syncthreads();

    for (int kk = 0; kk < K; kk += 32) {
        const int cur = (kk >> 5) & 1;
        if (kk + 32 < K) issue(cur ^ 1, kk + 32);   // overlap next tile with math

        v16bf af[2], bfr[4];
        #pragma unroll
        for (int mt = 0; mt < 2; ++mt) af[mt]  = load_A_frag(&As[cur][wm + mt * 16][0], 32, lane);
        #pragma unroll
        for (int nt = 0; nt < 4; ++nt) bfr[nt] = load_B_frag(&Ws[cur][wn + nt * 16][0], 32, lane);
        #pragma unroll
        for (int mt = 0; mt < 2; ++mt)
            #pragma unroll
            for (int nt = 0; nt < 4; ++nt)
                acc[mt][nt] = wmma_bf16(af[mt], bfr[nt], acc[mt][nt]);

        wait_async0();      // own async writes to buf cur^1 landed
        __syncthreads();    // everyone's landed; everyone done reading buf cur
    }

    // epilogue: C/D layout -> row = r + 8*(lane>=16), col = lane&15
    const int rhalf = (lane >> 4) * 8;
    const int nl    = lane & 15;
    #pragma unroll
    for (int mt = 0; mt < 2; ++mt) {
        #pragma unroll
        for (int nt = 0; nt < 4; ++nt) {
            #pragma unroll
            for (int r = 0; r < 8; ++r) {
                int m = m0 + wm + mt * 16 + r + rhalf;
                int n = n0 + wn + nt * 16 + nl;
                float v = acc[mt][nt][r] + bias[n];
                if constexpr (MODE == 0) {
                    outF[(size_t)m * C_ + n] = v;
                } else {
                    int b = m >> 11, t = m & (T_ - 1);
                    int h = n >> 6,  dd = n & (D_ - 1);
                    outQ[((((size_t)b * H_ + h) * T_) + t) * D_ + dd] = f2bf(v);
                }
            }
        }
    }
}

// ---------------- flash attention: per (b,h), 128 query rows / block ----------------
__global__ __launch_bounds__(256)
void attn_bf16(const __bf16* __restrict__ Q, const __bf16* __restrict__ Kx,
               const __bf16* __restrict__ V, __bf16* __restrict__ Y) {
    __shared__ __align__(64) __bf16 Ks[32][D_];      // 32 keys x 64 (4 KB)
    __shared__ __align__(64) __bf16 Vt[D_][32];      // transposed V tile (4 KB)
    __shared__ __align__(64) __bf16 Ps[8][16][32];   // per-wave P scratch (8 KB)

    const int bh = blockIdx.y;          // b*H + h
    const int q0 = blockIdx.x * 128;
    const __bf16* Qb = Q  + (size_t)bh * T_ * D_;
    const __bf16* Kb = Kx + (size_t)bh * T_ * D_;
    const __bf16* Vb = V  + (size_t)bh * T_ * D_;

    const int tid  = threadIdx.x;
    const int w    = tid >> 5;
    const int lane = tid & 31;
    const int qw   = q0 + w * 16;       // this wave's 16 query rows

    // Q fragments (16 x 64 = two 16x32 A fragments), loaded once
    v16bf qa[2];
    {
        int m  = lane & 15;
        int kh = (lane >> 4) * 8;
        const __bf16* qp = Qb + (size_t)(qw + m) * D_;
        #pragma unroll
        for (int i = 0; i < 2; ++i) {
            v16bf a;
            *(v8bf*)&a         = *(const v8bf*)(qp + i * 32 + kh);
            *(((v8bf*)&a) + 1) = *(const v8bf*)(qp + i * 32 + 16 + kh);
            qa[i] = a;
        }
    }

    const v8f vz = {0.f,0.f,0.f,0.f,0.f,0.f,0.f,0.f};
    v8f acc[4]; // O accumulator: 16 rows x 64 cols
    #pragma unroll
    for (int nd = 0; nd < 4; ++nd) acc[nd] = vz;
    float m_s[8], l_s[8];
    #pragma unroll
    for (int r = 0; r < 8; ++r) { m_s[r] = -1e30f; l_s[r] = 0.f; }

    const int rhalf = (lane >> 4) * 8;
    const int nl    = lane & 15;
    const int kmax  = q0 + 128;         // causal bound for this block (<= T)

    const int srow = tid >> 3;          // staging row (key) for this thread
    const int scol = (tid & 7) * 8;     // staging col (d) for this thread

    for (int kt = 0; kt < kmax; kt += 32) {
        // K tile: async global->LDS; V tile: registers + transposed scatter
        async_b128((unsigned)(size_t)&Ks[srow][scol],
                   (unsigned)(((kt + srow) * D_ + scol) * 2),
                   (unsigned long long)Kb);
        {
            v8bf vv = *(const v8bf*)(Vb + (size_t)(kt + srow) * D_ + scol);
            #pragma unroll
            for (int j = 0; j < 8; ++j) Vt[scol + j][srow] = vv[j];
        }
        if (kt + 32 < kmax) {   // warm L2/WGP$ for the next tile
            __builtin_prefetch(Kb + (size_t)(kt + 32 + srow) * D_ + scol, 0, 0);
            __builtin_prefetch(Vb + (size_t)(kt + 32 + srow) * D_ + scol, 0, 0);
        }
        wait_async0();
        __syncthreads();

        if (kt <= qw + 15) {            // wave has live keys in this tile
            // S = Q * K^T : 16 rows x 32 keys = two C tiles
            v8f s[2]; s[0] = vz; s[1] = vz;
            #pragma unroll
            for (int n0 = 0; n0 < 2; ++n0)
                #pragma unroll
                for (int kd = 0; kd < 2; ++kd)
                    s[n0] = wmma_bf16(qa[kd], load_B_frag(&Ks[n0 * 16][kd * 32], D_, lane), s[n0]);

            const float scale = 0.125f; // 1/sqrt(64)
            #pragma unroll
            for (int r = 0; r < 8; ++r) {
                int qrow = qw + r + rhalf;
                float v0 = (kt + nl      <= qrow) ? s[0][r] * scale : -1e30f;
                float v1 = (kt + 16 + nl <= qrow) ? s[1][r] * scale : -1e30f;
                float mx = fmaxf(v0, v1);
                mx = fmaxf(mx, __shfl_xor(mx, 1, 32));
                mx = fmaxf(mx, __shfl_xor(mx, 2, 32));
                mx = fmaxf(mx, __shfl_xor(mx, 4, 32));
                mx = fmaxf(mx, __shfl_xor(mx, 8, 32));
                float nm = fmaxf(m_s[r], mx);
                float al = __expf(m_s[r] - nm);
                m_s[r] = nm;
                float p0 = __expf(v0 - nm);
                float p1 = __expf(v1 - nm);
                float rs = p0 + p1;
                rs += __shfl_xor(rs, 1, 32);
                rs += __shfl_xor(rs, 2, 32);
                rs += __shfl_xor(rs, 4, 32);
                rs += __shfl_xor(rs, 8, 32);
                l_s[r] = l_s[r] * al + rs;
                #pragma unroll
                for (int nd = 0; nd < 4; ++nd) acc[nd][r] *= al;
                int prow = r + rhalf;
                Ps[w][prow][nl]      = f2bf(p0);
                Ps[w][prow][16 + nl] = f2bf(p1);
            }
            asm volatile("s_wait_dscnt 0x0" ::: "memory"); // LDS RAW (C->A relayout), same wave

            // P as A-fragment, then O += P * V
            v16bf pa;
            {
                int m  = lane & 15;
                int kh = (lane >> 4) * 8;
                const __bf16* pp = &Ps[w][m][0];
                *(v8bf*)&pa         = *(const v8bf*)(pp + kh);
                *(((v8bf*)&pa) + 1) = *(const v8bf*)(pp + 16 + kh);
            }
            #pragma unroll
            for (int nd = 0; nd < 4; ++nd)
                acc[nd] = wmma_bf16(pa, load_B_frag(&Vt[nd * 16][0], 32, lane), acc[nd]);
        }
        __syncthreads();
    }

    // epilogue: y[b, t, h*64+dc] = O / l
    const int b = bh >> 4, h = bh & (H_ - 1);
    #pragma unroll
    for (int r = 0; r < 8; ++r) {
        float inv = 1.0f / l_s[r];
        int qrow = qw + r + rhalf;
        size_t base = ((size_t)b * T_ + qrow) * C_ + h * D_ + nl;
        #pragma unroll
        for (int nd = 0; nd < 4; ++nd)
            Y[base + nd * 16] = f2bf(acc[nd][r] * inv);
    }
}

// ---------------- launcher ----------------
extern "C" void kernel_launch(void* const* d_in, const int* in_sizes, int n_in,
                              void* d_out, int out_size, void* d_ws, size_t ws_size,
                              hipStream_t stream) {
    const float* x  = (const float*)d_in[0];
    const float* Wq = (const float*)d_in[1];
    const float* bq = (const float*)d_in[2];
    const float* Wk = (const float*)d_in[3];
    const float* bk = (const float*)d_in[4];
    const float* Wv = (const float*)d_in[5];
    const float* bv = (const float*)d_in[6];
    const float* Wp = (const float*)d_in[7];
    const float* bp = (const float*)d_in[8];
    float* out = (float*)d_out;

    char* ws = (char*)d_ws;
    __bf16* xbf = (__bf16*)ws;  ws += (size_t)M_ * C_ * 2;
    __bf16* wqb = (__bf16*)ws;  ws += (size_t)C_ * C_ * 2;
    __bf16* wkb = (__bf16*)ws;  ws += (size_t)C_ * C_ * 2;
    __bf16* wvb = (__bf16*)ws;  ws += (size_t)C_ * C_ * 2;
    __bf16* wpb = (__bf16*)ws;  ws += (size_t)C_ * C_ * 2;
    __bf16* qb  = (__bf16*)ws;  ws += (size_t)M_ * C_ * 2;
    __bf16* kb  = (__bf16*)ws;  ws += (size_t)M_ * C_ * 2;
    __bf16* vb  = (__bf16*)ws;  ws += (size_t)M_ * C_ * 2;
    __bf16* yb  = (__bf16*)ws;  ws += (size_t)M_ * C_ * 2;

    // 1) convert inputs to bf16 (vectorized, bandwidth-bound)
    cvt_f32_bf16_v4<<<2048, 256, 0, stream>>>(x,  xbf, M_ * C_ / 4);
    cvt_f32_bf16_v4<<<512,  256, 0, stream>>>(Wq, wqb, C_ * C_ / 4);
    cvt_f32_bf16_v4<<<512,  256, 0, stream>>>(Wk, wkb, C_ * C_ / 4);
    cvt_f32_bf16_v4<<<512,  256, 0, stream>>>(Wv, wvb, C_ * C_ / 4);
    cvt_f32_bf16_v4<<<512,  256, 0, stream>>>(Wp, wpb, C_ * C_ / 4);

    // 2) QKV projections (bias fused, scattered to [B,H,T,d] bf16)
    dim3 gg(C_ / 128, M_ / 128);   // (8, 64)
    gemm_bf16<1><<<gg, 256, 0, stream>>>(xbf, wqb, bq, nullptr, qb);
    gemm_bf16<1><<<gg, 256, 0, stream>>>(xbf, wkb, bk, nullptr, kb);
    gemm_bf16<1><<<gg, 256, 0, stream>>>(xbf, wvb, bv, nullptr, vb);

    // 3) causal flash attention -> y [B,T,C] bf16
    dim3 ga(T_ / 128, B_ * H_);    // (16, 64)
    attn_bf16<<<ga, 256, 0, stream>>>(qb, kb, vb, yb);

    // 4) output projection -> f32 d_out (bias fused)
    gemm_bf16<0><<<gg, 256, 0, stream>>>(yb, wpb, bp, out, nullptr);
}
